// SelfAttention_78864189489197
// MI455X (gfx1250) — matrix operand order repaired and checked by reading, not compiled, not run on hardware
//
#include <hip/hip_runtime.h>
#include <hip/hip_bf16.h>

// MI455X / gfx1250 self-attention (q=k=v=x@Wq^T), f16 WMMA with f32 accum.
// Compute-bound (~146 GFLOP vs ~65MB traffic) -> everything through
// v_wmma_f32_16x16x32_f16. Workspace holds q in BOTH row-major and
// transposed layouts so every WMMA operand is a contiguous 128-bit load.
// GEMM loops are software-pipelined: iteration k issues iteration k+1's
// operand loads before consuming iteration k's with WMMAs, so waits before
// the WMMAs never have to drain to loadcnt==0.

typedef __attribute__((ext_vector_type(16))) _Float16 v16h;
typedef __attribute__((ext_vector_type(8)))  _Float16 v8h;
typedef __attribute__((ext_vector_type(8)))  float    v8f;

#define HD    512
#define SEQ   4096
#define NBAT  4
#define CHUNK 512
#define ATTN_SCALE 0.04419417382415922f   // 512^-0.5

__device__ __forceinline__ v16h mk16(v8h lo, v8h hi) {
    v16h r;
#pragma unroll
    for (int i = 0; i < 8; ++i) { r[i] = lo[i]; r[i + 8] = hi[i]; }
    return r;
}

__device__ __forceinline__ v8f wmma_f16(v16h a, v16h b, v8f c) {
    return __builtin_amdgcn_wmma_f32_16x16x32_f16(
        /*neg_a=*/false, a, /*neg_b=*/false, b,
        /*c_mod=*/(short)0, c, /*reuse_a=*/false, /*reuse_b=*/false);
}

// ---------------------------------------------------------------------------
// Kernel 0: one-time W f32 -> f16 (512x512, row-major [out][in])
// ---------------------------------------------------------------------------
__global__ __launch_bounds__(256)
void cvtw_kernel(const float* __restrict__ W, _Float16* __restrict__ Wh) {
    const int i = (blockIdx.x * 256 + threadIdx.x) * 4;
    float4 w = *(const float4*)(W + i);
    _Float16 h[4] = { (_Float16)w.x, (_Float16)w.y, (_Float16)w.z, (_Float16)w.w };
    *(unsigned long long*)(Wh + i) = *(unsigned long long*)h;  // b64 store
}

// ---------------------------------------------------------------------------
// Kernel 1: qh[r][d] = sum_k x[r][k] * Wh[d][k]; also writes transposed
// qhT[b][d][c]. grid = 1024 blocks x 256 threads (8 waves); block owns 16
// rows, wave w owns output cols [64w, 64w+64).
// ---------------------------------------------------------------------------
__global__ __launch_bounds__(256)
void proj_kernel(const float* __restrict__ x, const _Float16* __restrict__ Wh,
                 _Float16* __restrict__ qh, _Float16* __restrict__ qhT) {
    __shared__ __align__(16) _Float16 xh[16][HD];   // 16 KB staged x tile
    const int tid  = threadIdx.x;
    const int row0 = blockIdx.x * 16;

    for (int i = tid; i < 16 * HD; i += 256) {
        xh[i >> 9][i & (HD - 1)] =
            (_Float16)x[(size_t)(row0 + (i >> 9)) * HD + (i & (HD - 1))];
    }
    __syncthreads();

    const int wave = tid >> 5, lane = tid & 31;
    const int n16 = lane & 15, h = lane >> 4;

    // A: lane = row n16, K window {kb+16h .. kb+16h+15}
    auto ldA = [&](int kb) -> v16h {
        return mk16(*(const v8h*)&xh[n16][kb + 16 * h],
                    *(const v8h*)&xh[n16][kb + 16 * h + 8]);
    };
    // B[k][n] = Wh[col][k] -> lane n reads a contiguous Wh row
    auto ldB = [&](int tt, int kb) -> v16h {
        const _Float16* wr =
            Wh + (size_t)(wave * 64 + tt * 16 + n16) * HD + kb + 16 * h;
        return mk16(*(const v8h*)wr, *(const v8h*)(wr + 8));
    };

    v8f acc[4] = {};
    v16h A  = ldA(0);
    v16h B0 = ldB(0, 0), B1 = ldB(1, 0), B2 = ldB(2, 0), B3 = ldB(3, 0);
    for (int kk = 0; kk < HD / 32 - 1; ++kk) {
        const int kb = (kk + 1) * 32;
        v16h An  = ldA(kb);
        v16h N0 = ldB(0, kb), N1 = ldB(1, kb), N2 = ldB(2, kb), N3 = ldB(3, kb);
        acc[0] = wmma_f16(A, B0, acc[0]);
        acc[1] = wmma_f16(A, B1, acc[1]);
        acc[2] = wmma_f16(A, B2, acc[2]);
        acc[3] = wmma_f16(A, B3, acc[3]);
        A = An; B0 = N0; B1 = N1; B2 = N2; B3 = N3;
    }
    acc[0] = wmma_f16(A, B0, acc[0]);
    acc[1] = wmma_f16(A, B1, acc[1]);
    acc[2] = wmma_f16(A, B2, acc[2]);
    acc[3] = wmma_f16(A, B3, acc[3]);

#pragma unroll
    for (int tt = 0; tt < 4; ++tt) {
        const int col = wave * 64 + tt * 16 + n16;
        _Float16 hv[8];
#pragma unroll
        for (int g = 0; g < 8; ++g) {
            hv[g] = (_Float16)acc[tt][g];
            // row-major: D layout row = g + 8h, col = lane&15
            qh[(size_t)(row0 + g + 8 * h) * HD + col] = hv[g];
        }
        // transposed: rows g..g+7 are CONTIGUOUS along c -> single b128 store
        const int b = row0 >> 12;                   // batch of this row tile
        const int c = (row0 & (SEQ - 1)) + 8 * h;   // first seq index
        *(v8h*)(qhT + ((size_t)b * HD + col) * SEQ + c) = *(v8h*)hv;
    }
}

// ---------------------------------------------------------------------------
// Kernel 2: fused attention. One block per (batch, 16-row query tile).
// Online softmax over kv chunks of 512; O tile (16x512 f32) lives in LDS.
// ---------------------------------------------------------------------------
__global__ __launch_bounds__(256)
void attn_kernel(const _Float16* __restrict__ qh,
                 const _Float16* __restrict__ qhT, float* __restrict__ out) {
    __shared__ __align__(16) float    sO[16][HD];     // 32 KB output accum
    __shared__ __align__(16) _Float16 sP[16][CHUNK];  // 16 KB softmax probs
    __shared__ float s_m[16], s_l[16], s_alpha[16];
    __shared__ float s_rmax[8][16], s_rsum[8][16];

    const int tid = threadIdx.x;
    const int b   = blockIdx.x >> 8;          // 256 query tiles per batch
    const int q0  = (blockIdx.x & 255) * 16;
    const size_t bOff = (size_t)b * SEQ;

    if (tid < 16) { s_m[tid] = -1e30f; s_l[tid] = 0.f; }
    {
        float* Of = &sO[0][0];
        for (int i = tid; i < 16 * HD; i += 256) Of[i] = 0.f;
    }
    __syncthreads();

    const int wave = tid >> 5, lane = tid & 31;
    const int n16 = lane & 15, h = lane >> 4;

    for (int c0 = 0; c0 < SEQ; c0 += CHUNK) {
        // ---- Phase 1: S = scale * (Q K^T), wave w covers kv [c0+64w, +64) ----
        auto ldQ = [&](int kb) -> v16h {
            const _Float16* qa = qh + (bOff + q0 + n16) * HD + kb;
            return mk16(*(const v8h*)(qa + 16 * h),
                        *(const v8h*)(qa + 16 * h + 8));
        };
        // B[k][n] = K[kv+n][k] = qh row, contiguous in k
        auto ldK = [&](int tt, int kb) -> v16h {
            const _Float16* kp =
                qh + (bOff + c0 + wave * 64 + tt * 16 + n16) * HD + kb + 16 * h;
            return mk16(*(const v8h*)kp, *(const v8h*)(kp + 8));
        };

        v8f S[4] = {};
        {
            v16h A  = ldQ(0);
            v16h B0 = ldK(0, 0), B1 = ldK(1, 0), B2 = ldK(2, 0), B3 = ldK(3, 0);
            for (int kk = 0; kk < HD / 32 - 1; ++kk) {
                const int kb = (kk + 1) * 32;
                v16h An = ldQ(kb);
                v16h N0 = ldK(0, kb), N1 = ldK(1, kb), N2 = ldK(2, kb), N3 = ldK(3, kb);
                S[0] = wmma_f16(A, B0, S[0]);
                S[1] = wmma_f16(A, B1, S[1]);
                S[2] = wmma_f16(A, B2, S[2]);
                S[3] = wmma_f16(A, B3, S[3]);
                A = An; B0 = N0; B1 = N1; B2 = N2; B3 = N3;
            }
            S[0] = wmma_f16(A, B0, S[0]);
            S[1] = wmma_f16(A, B1, S[1]);
            S[2] = wmma_f16(A, B2, S[2]);
            S[3] = wmma_f16(A, B3, S[3]);
        }
        float rmax[8];
#pragma unroll
        for (int g = 0; g < 8; ++g) rmax[g] = -1e30f;
#pragma unroll
        for (int tt = 0; tt < 4; ++tt)
#pragma unroll
            for (int g = 0; g < 8; ++g) {
                S[tt][g] *= ATTN_SCALE;
                rmax[g] = fmaxf(rmax[g], S[tt][g]);
            }
        // row max across the 16 lanes of each half (rows g+8h)
#pragma unroll
        for (int g = 0; g < 8; ++g) {
            float v = rmax[g];
            v = fmaxf(v, __shfl_xor(v, 1, 16));
            v = fmaxf(v, __shfl_xor(v, 2, 16));
            v = fmaxf(v, __shfl_xor(v, 4, 16));
            v = fmaxf(v, __shfl_xor(v, 8, 16));
            if (n16 == 0) s_rmax[wave][g + 8 * h] = v;
        }
        __syncthreads();

        // ---- Phase 2: new running max / rescale factor ----
        if (tid < 16) {
            float mc = s_rmax[0][tid];
#pragma unroll
            for (int w2 = 1; w2 < 8; ++w2) mc = fmaxf(mc, s_rmax[w2][tid]);
            const float mold = s_m[tid];
            const float mnew = fmaxf(mold, mc);
            const float a    = __expf(mold - mnew);
            s_alpha[tid] = a;
            s_m[tid]     = mnew;
            s_l[tid]    *= a;
        }
        __syncthreads();

        // ---- Phase 3: O *= alpha; P = exp(S - m) -> LDS; row sums ----
        {
            float* Of = &sO[0][0];
            for (int i = tid; i < 16 * HD; i += 256) Of[i] *= s_alpha[i >> 9];
        }
        float mrow[8], rsum[8];
#pragma unroll
        for (int g = 0; g < 8; ++g) { mrow[g] = s_m[g + 8 * h]; rsum[g] = 0.f; }
#pragma unroll
        for (int tt = 0; tt < 4; ++tt) {
            const int pc = wave * 64 + tt * 16 + n16;
#pragma unroll
            for (int g = 0; g < 8; ++g) {
                const float p = __expf(S[tt][g] - mrow[g]);
                rsum[g] += p;
                sP[g + 8 * h][pc] = (_Float16)p;
            }
        }
#pragma unroll
        for (int g = 0; g < 8; ++g) {
            float v = rsum[g];
            v += __shfl_xor(v, 1, 16);
            v += __shfl_xor(v, 2, 16);
            v += __shfl_xor(v, 4, 16);
            v += __shfl_xor(v, 8, 16);
            if (n16 == 0) s_rsum[wave][g + 8 * h] = v;
        }
        __syncthreads();

        // ---- Phase 4: accumulate l (disjoint from phase 5's sO/sP work) ----
        if (tid < 16) {
            float s = 0.f;
#pragma unroll
            for (int w2 = 0; w2 < 8; ++w2) s += s_rsum[w2][tid];
            s_l[tid] += s;
        }

        // ---- Phase 5: O += P @ V, wave w owns exclusive d-slice [64w, +64) ----
        auto ldP = [&](int kb) -> v16h {
            return mk16(*(const v8h*)&sP[n16][kb + 16 * h],
                        *(const v8h*)&sP[n16][kb + 16 * h + 8]);
        };
        // B[k][n] = V[c0+kb+k][col] = qhT[col][c0+kb+k], contiguous in k
        auto ldV = [&](int tt, int kb) -> v16h {
            const _Float16* vp =
                qhT + ((size_t)b * HD + wave * 64 + tt * 16 + n16) * SEQ +
                c0 + kb + 16 * h;
            return mk16(*(const v8h*)vp, *(const v8h*)(vp + 8));
        };

        v8f acc[4] = {};
        {
            v16h A  = ldP(0);
            v16h B0 = ldV(0, 0), B1 = ldV(1, 0), B2 = ldV(2, 0), B3 = ldV(3, 0);
            for (int kk = 0; kk < CHUNK / 32 - 1; ++kk) {
                const int kb = (kk + 1) * 32;
                v16h An = ldP(kb);
                v16h N0 = ldV(0, kb), N1 = ldV(1, kb), N2 = ldV(2, kb), N3 = ldV(3, kb);
                acc[0] = wmma_f16(A, B0, acc[0]);
                acc[1] = wmma_f16(A, B1, acc[1]);
                acc[2] = wmma_f16(A, B2, acc[2]);
                acc[3] = wmma_f16(A, B3, acc[3]);
                A = An; B0 = N0; B1 = N1; B2 = N2; B3 = N3;
            }
            acc[0] = wmma_f16(A, B0, acc[0]);
            acc[1] = wmma_f16(A, B1, acc[1]);
            acc[2] = wmma_f16(A, B2, acc[2]);
            acc[3] = wmma_f16(A, B3, acc[3]);
        }
#pragma unroll
        for (int tt = 0; tt < 4; ++tt) {
            const int col = wave * 64 + tt * 16 + n16;
#pragma unroll
            for (int g = 0; g < 8; ++g) sO[g + 8 * h][col] += acc[tt][g];
        }
        __syncthreads();
    }

    // ---- Epilogue: out = O / l ----
    for (int i = tid; i < 16 * HD; i += 256) {
        const int r = i >> 9, c = i & (HD - 1);
        out[(bOff + q0 + r) * HD + c] = sO[r][c] / s_l[r];
    }
}

// ---------------------------------------------------------------------------
extern "C" void kernel_launch(void* const* d_in, const int* in_sizes, int n_in,
                              void* d_out, int out_size, void* d_ws, size_t ws_size,
                              hipStream_t stream) {
    const float* x  = (const float*)d_in[0];  // [4, 4096, 512] f32
    const float* Wq = (const float*)d_in[1];  // [512, 512] f32
    float* out = (float*)d_out;               // [4, 4096, 512] f32

    // workspace layout: qh (16 MB) | qhT (16 MB) | Wh (0.5 MB)
    _Float16* qh  = (_Float16*)d_ws;
    _Float16* qhT = qh  + (size_t)NBAT * SEQ * HD;
    _Float16* Wh  = qhT + (size_t)NBAT * SEQ * HD;

    cvtw_kernel<<<dim3(HD * HD / 1024), dim3(256), 0, stream>>>(Wq, Wh);
    const int blocks = NBAT * SEQ / 16;       // 1024
    proj_kernel<<<dim3(blocks), dim3(256), 0, stream>>>(x, Wh, qh, qhT);
    attn_kernel<<<dim3(blocks), dim3(256), 0, stream>>>(qh, qhT, out);
}
